// QuantizationLayer_16475494548010
// MI455X (gfx1250) — matrix operand order repaired and checked by reading, not compiled, not run on hardware
//
#include <hip/hip_runtime.h>

typedef __attribute__((ext_vector_type(16))) _Float16 v16h;
typedef __attribute__((ext_vector_type(8)))  float    v8f;
typedef __attribute__((ext_vector_type(4)))  float    v4f;

#define K_DIM 4096
#define D_DIM 64
// block = 256 threads = 8 waves; 2 row-tiles (16 rows each) per block,
// 4 waves per tile each covering K/4 = 1024 of the K dimension.

__global__ __launch_bounds__(256)
void vq_onehot_wmma_gather(const float* __restrict__ enc,
                           const float* __restrict__ cb,
                           float* __restrict__ out) {
  __shared__ float red[8][32][8];   // per-wave partial D fragments (8 KB)

  const int lane = threadIdx.x & 31;
  const int wave = threadIdx.x >> 5;
  const int tib  = wave >> 2;       // tile within block: 0..1
  const int ksec = wave & 3;        // K section: 0..3
  const int row0 = (blockIdx.x * 2 + tib) * 16;
  const int half = lane >> 4;       // 16-bit A layout: lanes 0-15 / 16-31

  // This lane streams row (row0 + lane%16) of the one-hot matrix.
  const float* rowp = enc + (size_t)(row0 + (lane & 15)) * K_DIM;

  // B_iota[k, n] = k_local (0..31), exact in f16.
  // 16-bit B layout: elems 0..7 -> K = 8*half + 0..7, elems 8..15 -> K = 16 + 8*half + 0..7.
  v16h biota;
#pragma unroll
  for (int e = 0; e < 16; ++e) biota[e] = (_Float16)(float)(e < 8 ? 8 * half + e : 8 + 8 * half + e);

  v8f acc = {};  // accumulates idx[m] in every column (columns identical)

  const int kbase = ksec * (K_DIM / 4);
#pragma unroll 2
  for (int cc = 0; cc < (K_DIM / 4) / 32; ++cc) {
    const int k0 = kbase + cc * 32;

    // A fragment, 16-bit 16x32 layout:
    //   lane L (M = L%16): elems 0..7  -> K = k0 + 8*half + 0..7
    //                      elems 8..15 -> K = k0 + 16 + 8*half + 0..7
    const float* p = rowp + k0 + 8 * half;
    v4f p0 = __builtin_nontemporal_load((const v4f*)(p));
    v4f p1 = __builtin_nontemporal_load((const v4f*)(p + 4));
    v4f p2 = __builtin_nontemporal_load((const v4f*)(p + 16));
    v4f p3 = __builtin_nontemporal_load((const v4f*)(p + 20));

    v16h a;
#pragma unroll
    for (int e = 0; e < 4; ++e) {
      a[e]      = (_Float16)p0[e];   // exact: values are 0.0 or 1.0
      a[4 + e]  = (_Float16)p1[e];
      a[8 + e]  = (_Float16)p2[e];
      a[12 + e] = (_Float16)p3[e];
    }

    // B_base[k, n] = k0 (multiple of 32, <= 4064 -> exact in f16)
    const _Float16 hb = (_Float16)(float)k0;
    v16h bbase;
#pragma unroll
    for (int e = 0; e < 16; ++e) bbase[e] = hb;

    // acc[m,n] += sum_k A[m,k] * (k_local + k0)  ==  global index when hit
    acc = __builtin_amdgcn_wmma_f32_16x16x32_f16(
        false, a, false, biota, (short)0, acc, false, false);
    acc = __builtin_amdgcn_wmma_f32_16x16x32_f16(
        false, a, false, bbase, (short)0, acc, false, false);
  }

  // Cross-wave K reduction through LDS.
#pragma unroll
  for (int j = 0; j < 8; ++j) red[wave][lane][j] = acc[j];
  __syncthreads();

  if (ksec == 0) {  // leader wave of each tile (wave = tib*4)
    int iv[8];
#pragma unroll
    for (int j = 0; j < 8; ++j) {
      float s = red[wave][lane][j] + red[wave + 1][lane][j] +
                red[wave + 2][lane][j] + red[wave + 3][lane][j];
      iv[j] = (int)(s + 0.5f);  // exact integer in [0, 4095]
    }
    // D layout: VGPR j, lanes 0-15 -> M=j, lanes 16-31 -> M=8+j.
    // Column N=0 lives on lanes 0 and 16; broadcast row indices and gather.
#pragma unroll
    for (int r = 0; r < 16; ++r) {
      const int kr = __builtin_amdgcn_readlane(iv[r & 7], (r < 8) ? 0 : 16);
      const float2* src = (const float2*)(cb + (size_t)kr * D_DIM);
      float2* dst = (float2*)(out + (size_t)(row0 + r) * D_DIM);
      dst[lane] = src[lane];  // 64 floats = 32 lanes x float2, coalesced
    }
  }
}

extern "C" void kernel_launch(void* const* d_in, const int* in_sizes, int n_in,
                              void* d_out, int out_size, void* d_ws, size_t ws_size,
                              hipStream_t stream) {
  const float* enc = (const float*)d_in[0];   // [N, 4096] one-hot f32
  const float* cb  = (const float*)d_in[1];   // [4096, 64] f32
  float* out = (float*)d_out;                 // [N, 64] f32

  const int n_rows = in_sizes[0] / K_DIM;     // 65536
  const int blocks = n_rows / 32;             // 2 tiles of 16 rows per block
  vq_onehot_wmma_gather<<<blocks, 256, 0, stream>>>(enc, cb, out);
}
